// MultiHeadAttention_3083786518941
// MI455X (gfx1250) — matrix operand order repaired and checked by reading, not compiled, not run on hardware
//
#include <hip/hip_runtime.h>

typedef __attribute__((ext_vector_type(16))) __bf16 v16bf;
typedef __attribute__((ext_vector_type(8)))  __bf16 v8bf;
typedef __attribute__((ext_vector_type(4)))  __bf16 v4bf;
typedef __attribute__((ext_vector_type(8)))  float  v8f;
typedef __attribute__((ext_vector_type(4)))  float  v4f;

#define D_MODEL   1024
#define NUM_HEADS 16
#define HEAD_DIM  64
#define BATCH     2
#define SEQ       2048
#define MROWS     (BATCH * SEQ)

#define WMMA_BF16(a, b, c) \
  __builtin_amdgcn_wmma_f32_16x16x32_bf16(false, (a), false, (b), (short)0, (c), false, false)

static __device__ __forceinline__ v16bf mk16(v8bf lo, v8bf hi) {
  v16bf r;
#pragma unroll
  for (int e = 0; e < 8; e++) { r[e] = lo[e]; r[e + 8] = hi[e]; }
  return r;
}

// ---------------------------------------------------------------------------
// fp32 -> bf16 elementwise (activations). 8 elems/thread, b128 in / b128 out.
// ---------------------------------------------------------------------------
__global__ __launch_bounds__(256) void cvt_bf16(const float* __restrict__ in,
                                                __bf16* __restrict__ out, int n) {
  int i = (blockIdx.x * 256 + threadIdx.x) * 8;
  if (i >= n) return;
  v4f a = *(const v4f*)(in + i);
  v4f b = *(const v4f*)(in + i + 4);
  v8bf o;
#pragma unroll
  for (int e = 0; e < 4; e++) { o[e] = (__bf16)a[e]; o[e + 4] = (__bf16)b[e]; }
  *(v8bf*)(out + i) = o;
}

// ---------------------------------------------------------------------------
// W[in][out] fp32 -> Wt[out][in] bf16 (one-time transpose+convert so GEMM
// B-fragments become contiguous b128 loads). Reads coalesced across lanes.
// ---------------------------------------------------------------------------
__global__ __launch_bounds__(256) void wt_kernel(const float* __restrict__ W,
                                                 __bf16* __restrict__ Wt) {
  int tid = blockIdx.x * 256 + threadIdx.x;   // 262144 threads
  int n   = tid & (D_MODEL - 1);
  int k0  = (tid >> 10) * 4;
  v4bf o;
#pragma unroll
  for (int i = 0; i < 4; i++)
    o[i] = (__bf16)W[(size_t)(k0 + i) * D_MODEL + n];
  *(v4bf*)(Wt + (size_t)n * D_MODEL + k0) = o;
}

// ---------------------------------------------------------------------------
// GEMM: out = X(bf16)[4096,1024] @ Wt(bf16,[out][in])^T + bias
// One wave: 2 m-tiles x 64 n (8 accumulators); B fragment reused by 2 WMMAs.
//   mode 0: bf16 out, head-split [B,H,S,64]        (Q, K)
//   mode 1: bf16 out, head-split transposed [B,H,64,S]  (V)
//   mode 2: fp32 out, row-major [4096,1024]        (final projection)
// ---------------------------------------------------------------------------
__global__ __launch_bounds__(256) void gemm_kernel(const __bf16* __restrict__ X,
                                                   const __bf16* __restrict__ Wt,
                                                   const float* __restrict__ bias,
                                                   __bf16* __restrict__ outb,
                                                   float* __restrict__ outf,
                                                   int mode) {
  const int lane = threadIdx.x & 31;
  const int wv   = threadIdx.x >> 5;
  const int lm   = lane & 15;
  const int hi   = lane >> 4;
  const int offA = hi * 16;   // A-fragment: 16 contiguous K per half-wave
  const int offB = hi * 16;   // B-fragment: 16 contiguous K per half-wave

  const int m0 = (blockIdx.x * 8 + wv) * 32;   // two 16-row tiles
  const int n0 = blockIdx.y * 64;

  v8f acc[2][4];
#pragma unroll
  for (int i = 0; i < 2; i++)
#pragma unroll
    for (int j = 0; j < 4; j++)
#pragma unroll
      for (int r = 0; r < 8; r++) acc[i][j][r] = 0.0f;

  const __bf16* xr0 = X + (size_t)(m0 + lm) * D_MODEL;
  const __bf16* xr1 = X + (size_t)(m0 + 16 + lm) * D_MODEL;

  for (int k = 0; k < D_MODEL; k += 32) {
    v16bf a0 = mk16(*(const v8bf*)(xr0 + k + offA),
                    *(const v8bf*)(xr0 + k + 8 + offA));
    v16bf a1 = mk16(*(const v8bf*)(xr1 + k + offA),
                    *(const v8bf*)(xr1 + k + 8 + offA));
#pragma unroll
    for (int j = 0; j < 4; j++) {
      const __bf16* wr = Wt + (size_t)(n0 + j * 16 + lm) * D_MODEL + k + offB;
      v16bf b = mk16(*(const v8bf*)(wr), *(const v8bf*)(wr + 8));
      acc[0][j] = WMMA_BF16(a0, b, acc[0][j]);
      acc[1][j] = WMMA_BF16(a1, b, acc[1][j]);
    }
  }

#pragma unroll
  for (int i = 0; i < 2; i++) {
    const int mt    = m0 + i * 16;
    const int bb    = mt / SEQ;
    const int sbase = mt % SEQ;
#pragma unroll
    for (int j = 0; j < 4; j++) {
      const int   n  = n0 + j * 16 + lm;
      const float bv = bias[n];
      const int   h  = n >> 6;
      const int   d  = n & 63;
      if (mode == 1) {        // [B,H,64,S]: rows r -> consecutive s, one b128
        v8bf pk;
#pragma unroll
        for (int r = 0; r < 8; r++) pk[r] = (__bf16)(acc[i][j][r] + bv);
        __bf16* dst = outb + ((size_t)((bb * NUM_HEADS + h) * HEAD_DIM + d)) * SEQ
                           + sbase + hi * 8;
        *(v8bf*)dst = pk;
      } else if (mode == 0) { // [B,H,S,64]
#pragma unroll
        for (int r = 0; r < 8; r++) {
          const int s = sbase + r + 8 * hi;
          outb[((size_t)((bb * NUM_HEADS + h) * SEQ + s)) * HEAD_DIM + d] =
              (__bf16)(acc[i][j][r] + bv);
        }
      } else {                // fp32 row-major
#pragma unroll
        for (int r = 0; r < 8; r++)
          outf[(size_t)(mt + r + 8 * hi) * D_MODEL + n] = acc[i][j][r] + bv;
      }
    }
  }
}

// ---------------------------------------------------------------------------
// Flash attention (causal). One wave owns a 16-row q tile; loops 32-key blocks
// with online softmax. Scores: 4x WMMA; P built via LDS transpose (D-layout ->
// A-layout) guarded by s_wait_dscnt; P@V: 4x WMMA into 4 O accumulators.
// ---------------------------------------------------------------------------
__global__ __launch_bounds__(256) void attn_kernel(const __bf16* __restrict__ Q,
                                                   const __bf16* __restrict__ K,
                                                   const __bf16* __restrict__ Vt,
                                                   __bf16* __restrict__ O) {
  __shared__ __align__(16) __bf16 plds[8][16][32];

  const int lane = threadIdx.x & 31;
  const int wv   = threadIdx.x >> 5;
  const int lm   = lane & 15;
  const int hi   = lane >> 4;
  const int offA = hi * 16;
  const int offB = hi * 16;

  const int q0 = (blockIdx.x * 8 + wv) * 16;
  const int h  = blockIdx.y;
  const int bb = blockIdx.z;

  const __bf16* Qh = Q  + ((size_t)(bb * NUM_HEADS + h)) * SEQ * HEAD_DIM;
  const __bf16* Kh = K  + ((size_t)(bb * NUM_HEADS + h)) * SEQ * HEAD_DIM;
  const __bf16* Vh = Vt + ((size_t)(bb * NUM_HEADS + h)) * HEAD_DIM * SEQ;

  v16bf aq[2];
  {
    const __bf16* qr = Qh + (size_t)(q0 + lm) * HEAD_DIM;
#pragma unroll
    for (int c = 0; c < 2; c++)
      aq[c] = mk16(*(const v8bf*)(qr + c * 32 + offA),
                   *(const v8bf*)(qr + c * 32 + 8 + offA));
  }

  v8f oacc[4];
#pragma unroll
  for (int j = 0; j < 4; j++)
#pragma unroll
    for (int r = 0; r < 8; r++) oacc[j][r] = 0.0f;

  float mst[8], lst[8];
#pragma unroll
  for (int r = 0; r < 8; r++) { mst[r] = -1e30f; lst[r] = 0.0f; }

  const int nkb = (q0 + 47) >> 5;  // causal: key blocks covering [0, q0+16)
  for (int kbi = 0; kbi < nkb; kbi++) {
    const int kb = kbi * 32;

    v8f s[2];
#pragma unroll
    for (int t = 0; t < 2; t++) {
#pragma unroll
      for (int r = 0; r < 8; r++) s[t][r] = 0.0f;
      const __bf16* kr = Kh + (size_t)(kb + t * 16 + lm) * HEAD_DIM;
#pragma unroll
      for (int c = 0; c < 2; c++) {
        v16bf b = mk16(*(const v8bf*)(kr + c * 32 + offB),
                       *(const v8bf*)(kr + c * 32 + offB + 8));
        s[t] = WMMA_BF16(aq[c], b, s[t]);
      }
    }

    // online softmax; rows r+8*hi live in this 16-lane half
#pragma unroll
    for (int r = 0; r < 8; r++) {
      const int q  = q0 + r + 8 * hi;
      const int k0 = kb + lm;
      float x0 = (k0      <= q) ? s[0][r] * 0.125f : -1e30f;
      float x1 = (k0 + 16 <= q) ? s[1][r] * 0.125f : -1e30f;
      float rm = fmaxf(x0, x1);
      rm = fmaxf(rm, __shfl_xor(rm, 1));
      rm = fmaxf(rm, __shfl_xor(rm, 2));
      rm = fmaxf(rm, __shfl_xor(rm, 4));
      rm = fmaxf(rm, __shfl_xor(rm, 8));
      const float mold = mst[r];
      const float mnew = fmaxf(mold, rm);
      const float al   = __expf(mold - mnew);
      const float p0   = __expf(x0 - mnew);
      const float p1   = __expf(x1 - mnew);
      float rs = p0 + p1;
      rs += __shfl_xor(rs, 1);
      rs += __shfl_xor(rs, 2);
      rs += __shfl_xor(rs, 4);
      rs += __shfl_xor(rs, 8);
      lst[r] = lst[r] * al + rs;
      mst[r] = mnew;
#pragma unroll
      for (int j = 0; j < 4; j++) oacc[j][r] *= al;
      const int prow = r + 8 * hi;
      plds[wv][prow][lm]      = (__bf16)p0;
      plds[wv][prow][lm + 16] = (__bf16)p1;
    }

    // LDS is in-order per wave; wait for stores, block compiler reordering.
    asm volatile("s_wait_dscnt 0" ::: "memory");

    v16bf pa;
    {
      const __bf16* pr = &plds[wv][lm][0];
      pa = mk16(*(const v8bf*)(pr + offA), *(const v8bf*)(pr + 8 + offA));
    }

#pragma unroll
    for (int j = 0; j < 4; j++) {
      const __bf16* vr = Vh + (size_t)(j * 16 + lm) * SEQ + kb + offB;
      v16bf b = mk16(*(const v8bf*)(vr), *(const v8bf*)(vr + 8));
      oacc[j] = WMMA_BF16(pa, b, oacc[j]);
    }
  }

#pragma unroll
  for (int r = 0; r < 8; r++) {
    const float inv = 1.0f / lst[r];
    const int   q   = q0 + r + 8 * hi;
#pragma unroll
    for (int j = 0; j < 4; j++) {
      O[((size_t)(bb * SEQ + q)) * D_MODEL + h * HEAD_DIM + j * 16 + lm] =
          (__bf16)(oacc[j][r] * inv);
    }
  }
}

// ---------------------------------------------------------------------------
extern "C" void kernel_launch(void* const* d_in, const int* in_sizes, int n_in,
                              void* d_out, int out_size, void* d_ws, size_t ws_size,
                              hipStream_t stream) {
  const float* query = (const float*)d_in[0];
  const float* key   = (const float*)d_in[1];
  const float* value = (const float*)d_in[2];
  // d_in[3] = mask (causal tril) — implemented analytically in attn_kernel
  const float* Wq = (const float*)d_in[4];
  const float* bq = (const float*)d_in[5];
  const float* Wk = (const float*)d_in[6];
  const float* bk = (const float*)d_in[7];
  const float* Wv = (const float*)d_in[8];
  const float* bv = (const float*)d_in[9];
  const float* Wo = (const float*)d_in[10];
  const float* bo = (const float*)d_in[11];

  const size_t XSZ = (size_t)MROWS * D_MODEL;    // 4 Mi elems
  const size_t WSZ = (size_t)D_MODEL * D_MODEL;  // 1 Mi elems
  __bf16* Xq  = (__bf16*)d_ws;
  __bf16* Xk  = Xq + XSZ;
  __bf16* Xv  = Xk + XSZ;
  __bf16* Qw  = Xv + XSZ;
  __bf16* Kw  = Qw + XSZ;
  __bf16* Vw  = Kw + XSZ;
  __bf16* Aw  = Vw + XSZ;
  __bf16* Wtq = Aw + XSZ;
  __bf16* Wtk = Wtq + WSZ;
  __bf16* Wtv = Wtk + WSZ;
  __bf16* Wto = Wtv + WSZ;   // total 64 MB of workspace

  dim3 blk(256);

  // one-time convert / transpose (everything stays L2-resident afterwards)
  const int ncvt = (int)(XSZ / 8 / 256);   // 2048 blocks
  cvt_bf16<<<ncvt, blk, 0, stream>>>(query, Xq, (int)XSZ);
  cvt_bf16<<<ncvt, blk, 0, stream>>>(key,   Xk, (int)XSZ);
  cvt_bf16<<<ncvt, blk, 0, stream>>>(value, Xv, (int)XSZ);
  const int nwt = (int)(WSZ / 4 / 256);    // 1024 blocks
  wt_kernel<<<nwt, blk, 0, stream>>>(Wq, Wtq);
  wt_kernel<<<nwt, blk, 0, stream>>>(Wk, Wtk);
  wt_kernel<<<nwt, blk, 0, stream>>>(Wv, Wtv);
  wt_kernel<<<nwt, blk, 0, stream>>>(Wo, Wto);

  // projections (WMMA GEMMs)
  dim3 g1(MROWS / 256, D_MODEL / 64, 1);   // 16 x 16; 8 waves x 2 m-tiles
  gemm_kernel<<<g1, blk, 0, stream>>>(Xq, Wtq, bq, Qw, nullptr, 0);
  gemm_kernel<<<g1, blk, 0, stream>>>(Xk, Wtk, bk, Kw, nullptr, 0);
  gemm_kernel<<<g1, blk, 0, stream>>>(Xv, Wtv, bv, Vw, nullptr, 1);

  attn_kernel<<<dim3(SEQ / 128, NUM_HEADS, BATCH), blk, 0, stream>>>(Qw, Kw, Vw, Aw);

  gemm_kernel<<<g1, blk, 0, stream>>>(Aw, Wto, bo, nullptr, (float*)d_out, 2);
}
